// HCHALayer_549755814399
// MI455X (gfx1250) — compile-verified, matrix operand order checked by reading, not run on hardware
//
#include <hip/hip_runtime.h>
#include <hip/hip_bf16.h>

#define N_NODES 50000
#define N_EDGES 20000
#define NNZV    800000
#define DIN     128
#define NH      4
#define DOUTV   64
#define HD      (NH * DOUTV)   // 256

typedef __attribute__((ext_vector_type(16))) __bf16 v16bf;
typedef __attribute__((ext_vector_type(8)))  float  v8f;

// ---------------------------------------------------------------------------
// One wave computes one 16x16 tile of (X @ W) via v_wmma_f32_16x16x32_bf16,
// then reduces it against attn (per-head dot) and atomically accumulates
// e_out[row, h].  M must be a multiple of 16 (50000 and 20000 both are).
// ---------------------------------------------------------------------------
__global__ void proj_attn_kernel(const float* __restrict__ X, int M,
                                 const float* __restrict__ W,     // [DIN, HD]
                                 const float* __restrict__ attn,  // [HD]
                                 float* __restrict__ eOut) {      // [M, NH]
  const int lane = threadIdx.x & 31;
  const int wave = blockIdx.x * (blockDim.x >> 5) + (threadIdx.x >> 5);
  const int totalTiles = (M >> 4) * (HD / 16);
  if (wave >= totalTiles) return;
  const int rt = wave >> 4;       // 16 column tiles
  const int ct = wave & 15;
  const int lo = lane & 15;
  const int hi = lane >> 4;

  const float* arow = X + (size_t)(rt * 16 + lo) * DIN;
  const int col = ct * 16 + lo;

  v8f c = {};
#pragma unroll
  for (int kk = 0; kk < DIN / 32; ++kk) {
    v16bf a, b;
#pragma unroll
    for (int i = 0; i < 8; ++i) {
      const int k = kk * 32 + ((i < 4) ? 0 : 16) + (i & 3) * 2 + hi * 8;
      float2 fa = *reinterpret_cast<const float2*>(arow + k);
      a[2 * i]     = (__bf16)fa.x;
      a[2 * i + 1] = (__bf16)fa.y;
      b[2 * i]     = (__bf16)W[(size_t)k * HD + col];
      b[2 * i + 1] = (__bf16)W[(size_t)(k + 1) * HD + col];
    }
    c = __builtin_amdgcn_wmma_f32_16x16x32_bf16(false, a, false, b,
                                                (short)0, c, false, false);
  }

  // attention-score reduction: e[row, h] += sum_n C[row, n] * attn[col(n)]
  const int h = ct >> 2;                 // 64 cols per head, 16 per tile
  const float aw = attn[ct * 16 + lo];
  float p[8];
#pragma unroll
  for (int r = 0; r < 8; ++r) p[r] = c[r] * aw;
#pragma unroll
  for (int off = 1; off < 16; off <<= 1) {
#pragma unroll
    for (int r = 0; r < 8; ++r) p[r] += __shfl_xor(p[r], off, 32);
  }
  if (lo == 0) {
    const int base = rt * 16 + hi * 8;   // C layout: M = r + 8*hi
#pragma unroll
    for (int r = 0; r < 8; ++r)
      atomicAdd(&eOut[(size_t)(base + r) * NH + h], p[r]);
  }
}

// ---------------------------------------------------------------------------
// ft = (vfeat @ W_v) * DV2[row], stored f32 to workspace [N, HD]
// ---------------------------------------------------------------------------
__global__ void proj_ft_kernel(const float* __restrict__ X,
                               const float* __restrict__ W,
                               const float* __restrict__ DV2,
                               float* __restrict__ ft) {
  const int lane = threadIdx.x & 31;
  const int wave = blockIdx.x * (blockDim.x >> 5) + (threadIdx.x >> 5);
  const int totalTiles = (N_NODES >> 4) * (HD / 16);
  if (wave >= totalTiles) return;
  const int rt = wave >> 4;
  const int ct = wave & 15;
  const int lo = lane & 15;
  const int hi = lane >> 4;

  const float* arow = X + (size_t)(rt * 16 + lo) * DIN;
  const int col = ct * 16 + lo;

  v8f c = {};
#pragma unroll
  for (int kk = 0; kk < DIN / 32; ++kk) {
    v16bf a, b;
#pragma unroll
    for (int i = 0; i < 8; ++i) {
      const int k = kk * 32 + ((i < 4) ? 0 : 16) + (i & 3) * 2 + hi * 8;
      float2 fa = *reinterpret_cast<const float2*>(arow + k);
      a[2 * i]     = (__bf16)fa.x;
      a[2 * i + 1] = (__bf16)fa.y;
      b[2 * i]     = (__bf16)W[(size_t)k * HD + col];
      b[2 * i + 1] = (__bf16)W[(size_t)(k + 1) * HD + col];
    }
    c = __builtin_amdgcn_wmma_f32_16x16x32_bf16(false, a, false, b,
                                                (short)0, c, false, false);
  }

#pragma unroll
  for (int r = 0; r < 8; ++r) {
    const int row = rt * 16 + hi * 8 + r;
    ft[(size_t)row * HD + ct * 16 + lo] = c[r] * DV2[row];
  }
}

// ---------------------------------------------------------------------------
// softmax pass 1: segment max over destination node (s >= 0, so int-ordered
// atomicMax on float bits is valid with a 0 init)
// ---------------------------------------------------------------------------
__global__ void attn_max_kernel(const int* __restrict__ ni, const int* __restrict__ ei,
                                const float* __restrict__ e_v, const float* __restrict__ e_e,
                                int* __restrict__ mmax) {
  const int i = blockIdx.x * blockDim.x + threadIdx.x;
  if (i >= NNZV) return;
  const int v = ni[i], e = ei[i];
#pragma unroll
  for (int h = 0; h < NH; ++h) {
    const float s = fmaxf(e_e[(size_t)e * NH + h] + e_v[(size_t)v * NH + h], 0.f);
    atomicMax(&mmax[(size_t)v * NH + h], __float_as_int(s));
  }
}

// softmax pass 2: denominator
__global__ void attn_den_kernel(const int* __restrict__ ni, const int* __restrict__ ei,
                                const float* __restrict__ e_v, const float* __restrict__ e_e,
                                const int* __restrict__ mmax, float* __restrict__ den) {
  const int i = blockIdx.x * blockDim.x + threadIdx.x;
  if (i >= NNZV) return;
  const int v = ni[i], e = ei[i];
#pragma unroll
  for (int h = 0; h < NH; ++h) {
    const float s = fmaxf(e_e[(size_t)e * NH + h] + e_v[(size_t)v * NH + h], 0.f);
    const float ex = __expf(s - __int_as_float(mmax[(size_t)v * NH + h]));
    atomicAdd(&den[(size_t)v * NH + h], ex);
  }
}

// ---------------------------------------------------------------------------
// stage 1 (nodes -> hyperedges), mean over H folded into the accumulation:
// eout[e,d] += sum_h ft[v, h*64+d] * a[i,h] / H.  64 threads per incidence;
// NNZ divisible by 4 so every block is full (syncthreads is safe).
// ---------------------------------------------------------------------------
__global__ void stage1_kernel(const int* __restrict__ ni, const int* __restrict__ ei,
                              const float* __restrict__ e_v, const float* __restrict__ e_e,
                              const int* __restrict__ mmax, const float* __restrict__ den,
                              const float* __restrict__ ft, float* __restrict__ eout) {
  __shared__ float aL[4][NH];
  const int g = threadIdx.x >> 6;
  const int d = threadIdx.x & 63;
  const int i = blockIdx.x * 4 + g;
  const int v = ni[i], e = ei[i];
  if (d < NH) {
    const int h = d;
    const float s = fmaxf(e_e[(size_t)e * NH + h] + e_v[(size_t)v * NH + h], 0.f);
    const float ex = __expf(s - __int_as_float(mmax[(size_t)v * NH + h]));
    aL[g][h] = 0.25f * ex / den[(size_t)v * NH + h];
  }
  __syncthreads();
  float acc = 0.f;
#pragma unroll
  for (int h = 0; h < NH; ++h)
    acc += ft[(size_t)v * HD + h * DOUTV + d] * aL[g][h];
  atomicAdd(&eout[(size_t)e * DOUTV + d], acc);
}

// ---------------------------------------------------------------------------
// stage 2 (hyperedges -> nodes): vout[v,d] += eout[e,d]*invDE[e] * mean_h(a)
// ---------------------------------------------------------------------------
__global__ void stage2_kernel(const int* __restrict__ ni, const int* __restrict__ ei,
                              const float* __restrict__ e_v, const float* __restrict__ e_e,
                              const int* __restrict__ mmax, const float* __restrict__ den,
                              const float* __restrict__ eout, const float* __restrict__ invDE,
                              float* __restrict__ vout) {
  __shared__ float aL[4][NH];
  const int g = threadIdx.x >> 6;
  const int d = threadIdx.x & 63;
  const int i = blockIdx.x * 4 + g;
  const int v = ni[i], e = ei[i];
  if (d < NH) {
    const float s = fmaxf(e_e[(size_t)e * NH + d] + e_v[(size_t)v * NH + d], 0.f);
    const float ex = __expf(s - __int_as_float(mmax[(size_t)v * NH + d]));
    aL[g][d] = 0.25f * ex / den[(size_t)v * NH + d];
  }
  __syncthreads();
  const float abar = aL[g][0] + aL[g][1] + aL[g][2] + aL[g][3];
  const float z = eout[(size_t)e * DOUTV + d] * invDE[e];
  atomicAdd(&vout[(size_t)v * DOUTV + d], z * abar);
}

__global__ void scale_out_kernel(float* __restrict__ vout, const float* __restrict__ DV2) {
  const int i = blockIdx.x * blockDim.x + threadIdx.x;
  if (i < N_NODES * DOUTV) vout[i] *= DV2[i >> 6];
}

// ---------------------------------------------------------------------------
extern "C" void kernel_launch(void* const* d_in, const int* in_sizes, int n_in,
                              void* d_out, int out_size, void* d_ws, size_t ws_size,
                              hipStream_t stream) {
  const float* vfeat  = (const float*)d_in[0];
  const float* efeat  = (const float*)d_in[1];
  const float* DV2    = (const float*)d_in[2];
  const float* invDE  = (const float*)d_in[3];
  const float* W_p    = (const float*)d_in[4];
  const float* W_v    = (const float*)d_in[5];
  const float* attn_v = (const float*)d_in[6];
  const float* attn_e = (const float*)d_in[7];
  const int*   nidx   = (const int*)d_in[8];
  const int*   eidx   = (const int*)d_in[9];

  float* out  = (float*)d_out;
  float* vout = out;                            // [N, 64]
  float* eout = out + (size_t)N_NODES * DOUTV;  // [E, 64]

  // workspace layout (~54 MB)
  float* ft  = (float*)d_ws;                            // N*256 f32
  float* e_v = ft + (size_t)N_NODES * HD;               // N*4
  float* e_e = e_v + (size_t)N_NODES * NH;              // E*4
  int*   mmx = (int*)(e_e + (size_t)N_EDGES * NH);      // N*4 (float bits)
  float* den = (float*)(mmx + (size_t)N_NODES * NH);    // N*4

  // zero accumulators (e_v, e_e, mmx, den are contiguous) and outputs
  hipMemsetAsync(e_v, 0,
                 ((size_t)N_NODES * NH * 3 + (size_t)N_EDGES * NH) * sizeof(float),
                 stream);
  hipMemsetAsync(d_out, 0, (size_t)(N_NODES + N_EDGES) * DOUTV * sizeof(float), stream);

  // projections + attention scores (WMMA bf16, f32 accumulate)
  {
    const int waves = (N_NODES / 16) * (HD / 16);       // 50000
    proj_attn_kernel<<<waves / 8, 256, 0, stream>>>(vfeat, N_NODES, W_p, attn_v, e_v);
  }
  {
    const int waves = (N_EDGES / 16) * (HD / 16);       // 20000
    proj_attn_kernel<<<waves / 8, 256, 0, stream>>>(efeat, N_EDGES, W_p, attn_e, e_e);
  }
  {
    const int waves = (N_NODES / 16) * (HD / 16);
    proj_ft_kernel<<<waves / 8, 256, 0, stream>>>(vfeat, W_v, DV2, ft);
  }

  // edge-softmax grouped by destination node
  attn_max_kernel<<<(NNZV + 255) / 256, 256, 0, stream>>>(nidx, eidx, e_v, e_e, mmx);
  attn_den_kernel<<<(NNZV + 255) / 256, 256, 0, stream>>>(nidx, eidx, e_v, e_e, mmx, den);

  // gather/scatter stages (mean over H folded in)
  stage1_kernel<<<NNZV / 4, 256, 0, stream>>>(nidx, eidx, e_v, e_e, mmx, den, ft, eout);
  stage2_kernel<<<NNZV / 4, 256, 0, stream>>>(nidx, eidx, e_v, e_e, mmx, den, eout, invDE, vout);
  scale_out_kernel<<<(N_NODES * DOUTV + 255) / 256, 256, 0, stream>>>(vout, DV2);
}
